// BmmEnsemble_77034533421748
// MI455X (gfx1250) — matrix operand order repaired and checked by reading, not compiled, not run on hardware
//
#include <hip/hip_runtime.h>

// ---------------------------------------------------------------------------
// MI455X (gfx1250) fused species-routed ensemble MLP
//   energy = sum_atoms mean_models MLP_{species(atom)}(aev[atom])
// f16 WMMA (v_wmma_f32_16x16x32_f16) with f32 accumulation.
// ---------------------------------------------------------------------------

typedef __attribute__((ext_vector_type(16))) _Float16 v16h;
typedef __attribute__((ext_vector_type(8)))  float    v8f;
typedef __attribute__((ext_vector_type(4)))  _Float16 h4;

#define N_SPECIES  4
#define NUM_MODELS 8
#define N_ATOMS    50000
#define NPER       (N_ATOMS / N_SPECIES)      // 12500 atoms per species
#define D0 384
#define D1 160
#define D2 128
#define D3 96
#define TILE 64                                // atoms per workgroup
#define NTILES ((NPER + TILE - 1) / TILE)      // 196
#define CELU_ALPHA 0.1f

// ---- workspace layout (bytes), all 256B aligned ---------------------------
#define WS_CNT   0u
#define WS_IDX   256u                                     // 4*12500*4 = 200000 B
#define WS_W1T   200448u                                  // f16 [S*M][160][384]
#define WS_W2T   (WS_W1T + 3932160u)                      // f16 [S*M][128][160]
#define WS_W3T   (WS_W2T + 1310720u)                      // f16 [S*M][ 96][128]
// end = WS_W3T + 786432 = 6,229,760 B (~6 MB of d_ws used)

// ---------------------------------------------------------------------------
__global__ void bmm_init_kernel(float* out, int* cnt) {
  if (threadIdx.x == 0) out[0] = 0.0f;
  if (threadIdx.x < N_SPECIES) cnt[threadIdx.x] = 0;
}

// Block-aggregated species routing: 4 global atomics per block.
__global__ void bmm_route_kernel(const int* __restrict__ species,
                                 int* __restrict__ cnt,
                                 int* __restrict__ idx) {
  __shared__ int localCnt[N_SPECIES];
  __shared__ int baseOff[N_SPECIES];
  const int i = blockIdx.x * 256 + threadIdx.x;
  if (threadIdx.x < N_SPECIES) localCnt[threadIdx.x] = 0;
  __syncthreads();
  int s = 0, p = 0;
  const bool valid = (i < N_ATOMS);
  if (valid) {
    s = species[i];
    p = atomicAdd(&localCnt[s], 1);     // LDS atomic
  }
  __syncthreads();
  if (threadIdx.x < N_SPECIES)
    baseOff[threadIdx.x] = atomicAdd(&cnt[threadIdx.x], localCnt[threadIdx.x]);
  __syncthreads();
  if (valid) idx[s * NPER + baseOff[s] + p] = i;
}

// f32 [mats][fin][fout]  ->  f16 [mats][fout][fin]   (W^T, B-fragment friendly)
__global__ void bmm_transpose_kernel(const float* __restrict__ src,
                                     _Float16* __restrict__ dst,
                                     int fin, int fout, int total) {
  const int e = blockIdx.x * 256 + threadIdx.x;
  if (e >= total) return;
  const int mf  = fin * fout;
  const int mat = e / mf;
  const int r   = e - mat * mf;
  const int f   = r / fout;
  const int o   = r - f * fout;
  dst[(size_t)mat * mf + (size_t)o * fin + f] = (_Float16)src[e];
}

// ---- WMMA fragment helpers (CDNA5 ISA 7.12.2 layouts) ---------------------
// A (16x32 f16): lane<16 -> row=lane, halves K0-7 & K16-23 ; lane>=16 -> row=lane-16, K8-15 & K24-31
__device__ __forceinline__ v16h loadA(const _Float16* H, int stride, int rowBase,
                                      int k0, int lane) {
  const int r  = rowBase + (lane & 15);
  const int hi = (lane >> 4) & 1;
  const _Float16* p = H + r * stride + k0 + hi * 8;
  union { v16h h; uint4 q[2]; } f;
  f.q[0] = *(const uint4*)(p);
  f.q[1] = *(const uint4*)(p + 16);
  return f.h;
}

// B (32x16 f16), W stored transposed [fout][fin]: lane<16 -> col=lane, K0-7 & K16-23 ; lane>=16 -> col=lane-16, K8-15 & K24-31
__device__ __forceinline__ v16h loadB(const _Float16* Wt, int fin, int colBase,
                                      int k0, int lane) {
  const int n  = colBase + (lane & 15);
  const int hi = (lane >> 4) & 1;
  const _Float16* p = Wt + (size_t)n * fin + k0 + hi * 8;
  union { v16h h; uint4 q[2]; } f;
  f.q[0] = *(const uint4*)(p);
  f.q[1] = *(const uint4*)(p + 16);
  return f.h;
}

__device__ __forceinline__ float celu_f(float x) {
  return x > 0.0f ? x : CELU_ALPHA * (__expf(x * (1.0f / CELU_ALPHA)) - 1.0f);
}

// C/D (16x16 f32): VGPR i, lane<16 -> M=i,N=lane ; lane>=16 -> M=i+8,N=lane-16
__device__ __forceinline__ void storeC(_Float16* H, int stride, int rowBase,
                                       int colBase, v8f acc, int lane, bool act) {
  const int col = colBase + (lane & 15);
  const int r0  = rowBase + ((lane >> 4) ? 8 : 0);
#pragma unroll
  for (int i = 0; i < 8; ++i) {
    float x = acc[i];
    if (act) x = celu_f(x);
    H[(r0 + i) * stride + col] = (_Float16)x;
  }
}

// One dense layer on a 64-atom tile: all 8 waves split the 16-col output tiles;
// each wave owns 4 row-tiles so every B fragment feeds 4 WMMAs.
// unroll_count(4) bounds the software-pipeline window so the wave stays within
// 256 VGPRs (no s_set_vgpr_msb thrash, >=4 waves/SIMD occupancy).
__device__ __forceinline__ void layer64(const _Float16* inH, int fin,
                                        _Float16* outH, int fout,
                                        const _Float16* Wt, const float* bias,
                                        int wave, int lane, bool act) {
  const int ncols = fout / 16;
  for (int c = wave; c < ncols; c += 8) {
    const float bv = bias[c * 16 + (lane & 15)];
    v8f a0 = {bv, bv, bv, bv, bv, bv, bv, bv};
    v8f a1 = a0, a2 = a0, a3 = a0;
#pragma clang loop unroll_count(4)
    for (int k0 = 0; k0 < fin; k0 += 32) {
      const v16h B  = loadB(Wt, fin, c * 16, k0, lane);
      const v16h A0 = loadA(inH, fin, 0,  k0, lane);
      const v16h A1 = loadA(inH, fin, 16, k0, lane);
      const v16h A2 = loadA(inH, fin, 32, k0, lane);
      const v16h A3 = loadA(inH, fin, 48, k0, lane);
      a0 = __builtin_amdgcn_wmma_f32_16x16x32_f16(false, A0, false, B, (short)0, a0, false, false);
      a1 = __builtin_amdgcn_wmma_f32_16x16x32_f16(false, A1, false, B, (short)0, a1, false, false);
      a2 = __builtin_amdgcn_wmma_f32_16x16x32_f16(false, A2, false, B, (short)0, a2, false, false);
      a3 = __builtin_amdgcn_wmma_f32_16x16x32_f16(false, A3, false, B, (short)0, a3, false, false);
    }
    storeC(outH, fout, 0,  c * 16, a0, lane, act);
    storeC(outH, fout, 16, c * 16, a1, lane, act);
    storeC(outH, fout, 32, c * 16, a2, lane, act);
    storeC(outH, fout, 48, c * 16, a3, lane, act);
  }
}

// ---------------------------------------------------------------------------
__global__ __launch_bounds__(256, 4)
void bmm_mlp_kernel(const float* __restrict__ aev,
                    const int* __restrict__ idx,
                    const _Float16* __restrict__ W1t,
                    const _Float16* __restrict__ W2t,
                    const _Float16* __restrict__ W3t,
                    const float* __restrict__ b1,
                    const float* __restrict__ b2,
                    const float* __restrict__ b3,
                    const float* __restrict__ W4,
                    const float* __restrict__ b4,
                    float* __restrict__ out) {
  const int s    = blockIdx.x / NTILES;
  const int t    = blockIdx.x % NTILES;
  const int base = t * TILE;
  const int tid  = threadIdx.x;
  const int lane = tid & 31;   // wave32 (gfx1250)
  const int wave = tid >> 5;

  __shared__ __attribute__((aligned(16))) _Float16 sX [TILE * D0];  // 48 KB
  __shared__ __attribute__((aligned(16))) _Float16 sH1[TILE * D1];  // 20 KB
  __shared__ __attribute__((aligned(16))) _Float16 sH2[TILE * D2];  // 16 KB
  __shared__ __attribute__((aligned(16))) _Float16 sH3[TILE * D3];  // 12 KB

  // ---- gather AEV tile (f32 -> f16), zero-fill tail rows ------------------
  const int* sidx = idx + s * NPER;
  for (int j = tid; j < TILE * (D0 / 4); j += 256) {
    const int row = j / (D0 / 4);
    const int c   = (j % (D0 / 4)) * 4;
    float4 v = make_float4(0.f, 0.f, 0.f, 0.f);
    const int ai = base + row;
    if (ai < NPER) {
      const float* p = aev + (size_t)sidx[ai] * D0 + c;
      v = *(const float4*)p;
    }
    h4 hv;
    hv[0] = (_Float16)v.x; hv[1] = (_Float16)v.y;
    hv[2] = (_Float16)v.z; hv[3] = (_Float16)v.w;
    *(h4*)(&sX[row * D0 + c]) = hv;
  }
  __syncthreads();

  float energy = 0.0f;  // per-thread (tid<64 -> atom base+tid), summed over models

  for (int m = 0; m < NUM_MODELS; ++m) {
    const int sm = s * NUM_MODELS + m;
    layer64(sX,  D0, sH1, D1, W1t + (size_t)sm * D1 * D0, b1 + sm * D1, wave, lane, true);
    __syncthreads();
    layer64(sH1, D1, sH2, D2, W2t + (size_t)sm * D2 * D1, b2 + sm * D2, wave, lane, true);
    __syncthreads();
    layer64(sH2, D2, sH3, D3, W3t + (size_t)sm * D3 * D2, b3 + sm * D3, wave, lane, true);
    __syncthreads();
    if (tid < TILE) {            // layer 4: 96 -> 1 dot product (VALU, scalar W loads)
      const float* w4 = W4 + sm * D3;
      const _Float16* h = &sH3[tid * D3];
      float e = b4[sm];
#pragma unroll
      for (int f = 0; f < D3; ++f) e += (float)h[f] * w4[f];
      energy += e;
    }
    __syncthreads();
  }

  if (tid < TILE && (base + tid) >= NPER) energy = 0.0f;  // mask tail atoms
  energy *= (1.0f / NUM_MODELS);                          // mean over models
#pragma unroll
  for (int off = 16; off > 0; off >>= 1) energy += __shfl_down(energy, off, 32);
  if (lane == 0 && wave < 2) atomicAdd(out, energy);
}

// ---------------------------------------------------------------------------
extern "C" void kernel_launch(void* const* d_in, const int* in_sizes, int n_in,
                              void* d_out, int out_size, void* d_ws, size_t ws_size,
                              hipStream_t stream) {
  (void)in_sizes; (void)n_in; (void)out_size; (void)ws_size;
  const int*   species = (const int*)d_in[0];
  const float* aev = (const float*)d_in[1];
  const float* W1  = (const float*)d_in[2];
  const float* b1  = (const float*)d_in[3];
  const float* W2  = (const float*)d_in[4];
  const float* b2  = (const float*)d_in[5];
  const float* W3  = (const float*)d_in[6];
  const float* b3  = (const float*)d_in[7];
  const float* W4  = (const float*)d_in[8];
  const float* b4  = (const float*)d_in[9];
  float* out = (float*)d_out;

  char* ws = (char*)d_ws;
  int*      cnt = (int*)(ws + WS_CNT);
  int*      idx = (int*)(ws + WS_IDX);
  _Float16* W1t = (_Float16*)(ws + WS_W1T);
  _Float16* W2t = (_Float16*)(ws + WS_W2T);
  _Float16* W3t = (_Float16*)(ws + WS_W3T);

  hipLaunchKernelGGL(bmm_init_kernel, dim3(1), dim3(64), 0, stream, out, cnt);
  hipLaunchKernelGGL(bmm_route_kernel, dim3((N_ATOMS + 255) / 256), dim3(256), 0,
                     stream, species, cnt, idx);
  const int t1 = N_SPECIES * NUM_MODELS * D0 * D1;
  const int t2 = N_SPECIES * NUM_MODELS * D1 * D2;
  const int t3 = N_SPECIES * NUM_MODELS * D2 * D3;
  hipLaunchKernelGGL(bmm_transpose_kernel, dim3((t1 + 255) / 256), dim3(256), 0,
                     stream, W1, W1t, D0, D1, t1);
  hipLaunchKernelGGL(bmm_transpose_kernel, dim3((t2 + 255) / 256), dim3(256), 0,
                     stream, W2, W2t, D1, D2, t2);
  hipLaunchKernelGGL(bmm_transpose_kernel, dim3((t3 + 255) / 256), dim3(256), 0,
                     stream, W3, W3t, D2, D3, t3);
  hipLaunchKernelGGL(bmm_mlp_kernel, dim3(N_SPECIES * NTILES), dim3(256), 0, stream,
                     aev, idx, W1t, W2t, W3t, b1, b2, b3, W4, b4, out);
}
